// SemanticAssignmentModule_45956150067385
// MI455X (gfx1250) — compile-verified, matrix operand order checked by reading, not run on hardware
//
#include <hip/hip_runtime.h>
#include <hip/hip_bf16.h>

// ---------------------------------------------------------------- types ----
typedef __bf16 bf16_t;
typedef __attribute__((ext_vector_type(16))) __bf16 v16bf;
typedef __attribute__((ext_vector_type(8)))  float  v8f;
typedef __attribute__((ext_vector_type(4)))  unsigned int v4u;

union FragBF { v16bf v; v4u q[2]; };

#define RSQRT_1P001 0.99950037f   // 1/sqrt(1.001)

// ------------------------------------------------ bf16 WMMA GEMM -----------
// D[M x N] (f32) = A[M x K] (bf16 row-major) * Bt[N x K]^T (bf16 row-major)
// epilogue: 0 = none, 1 = swish(x + bias[n])
// Block tile 128x128 (8 waves, each 64x32 = 4x2 fragments).
// K-slabs of 32 are staged global->LDS with async b128 copies (ASYNCcnt),
// double-buffered; WMMA operands come from LDS via ds_load_b128.
__global__ __launch_bounds__(256) void gemm_bf16_wmma(
    const bf16_t* __restrict__ A,  long aBS,
    const bf16_t* __restrict__ Bt, long bBS,
    float* __restrict__ D,         long dBS,
    int M, int N, int K,
    const float* __restrict__ bias, int epilogue)
{
  // LDS staging: [2 buffers][128 rows][32 bf16]  (8 KB per matrix per buffer)
  __shared__ __align__(16) bf16_t ldsA[2][128 * 32];
  __shared__ __align__(16) bf16_t ldsB[2][128 * 32];

  const int tid  = threadIdx.x;
  const int lane = tid & 31;
  const int wave = tid >> 5;
  const int wm = wave & 1;   // 2 waves along M
  const int wn = wave >> 1;  // 4 waves along N
  const int b  = blockIdx.z;

  const int mBlk = blockIdx.y * 128;
  const int nBlk = blockIdx.x * 128;

  A  += (long)b * aBS;
  Bt += (long)b * bBS;
  D  += (long)b * dBS;

  const int lrow  = lane & 15;        // M (for A) / N (for B) within fragment
  const int lksel = (lane >> 4) * 8;  // K sub-offset per lane half

  // per-thread async-copy sources: chunk id c = tid (+256); row = c>>2, 16B col = c&3
  const bf16_t* aSrc0 = A  + (long)(mBlk + (tid >> 2)) * K + (tid & 3) * 8;
  const bf16_t* aSrc1 = aSrc0 + (long)64 * K;
  const bf16_t* bSrc0 = Bt + (long)(nBlk + (tid >> 2)) * K + (tid & 3) * 8;
  const bf16_t* bSrc1 = bSrc0 + (long)64 * K;

  const unsigned ldsA_off = (unsigned)(size_t)&ldsA[0][0];  // generic->LDS: low 32 bits
  const unsigned ldsB_off = (unsigned)(size_t)&ldsB[0][0];
  const unsigned ABYTES = 128u * 32u * 2u;                  // 8 KB per buffer

  auto issue_stage = [&](int buf, int k0) {
    const unsigned da0 = ldsA_off + (unsigned)buf * ABYTES + (unsigned)tid * 16u;
    const unsigned da1 = da0 + 4096u;
    const unsigned db0 = ldsB_off + (unsigned)buf * ABYTES + (unsigned)tid * 16u;
    const unsigned db1 = db0 + 4096u;
    asm volatile("global_load_async_to_lds_b128 %0, %1, off"
                 :: "v"(da0), "v"((unsigned long long)(size_t)(aSrc0 + k0)) : "memory");
    asm volatile("global_load_async_to_lds_b128 %0, %1, off"
                 :: "v"(da1), "v"((unsigned long long)(size_t)(aSrc1 + k0)) : "memory");
    asm volatile("global_load_async_to_lds_b128 %0, %1, off"
                 :: "v"(db0), "v"((unsigned long long)(size_t)(bSrc0 + k0)) : "memory");
    asm volatile("global_load_async_to_lds_b128 %0, %1, off"
                 :: "v"(db1), "v"((unsigned long long)(size_t)(bSrc1 + k0)) : "memory");
  };

  v8f acc[4][2];
  const v8f vzero = {0.f,0.f,0.f,0.f,0.f,0.f,0.f,0.f};
  #pragma unroll
  for (int i = 0; i < 4; ++i)
    #pragma unroll
    for (int j = 0; j < 2; ++j) acc[i][j] = vzero;

  issue_stage(0, 0);  // prologue

  for (int k0 = 0; k0 < K; k0 += 32) {
    const int cur = (k0 >> 5) & 1;
    const bool has_next = (k0 + 32) < K;
    if (has_next) {
      issue_stage(cur ^ 1, k0 + 32);
      // pull the slab after next toward L2 while we're at it
      if (k0 + 64 < K)
        __builtin_prefetch((const void*)(aSrc0 + k0 + 64), 0, 1);
      asm volatile("s_wait_asynccnt 0x4" ::: "memory");  // stage k0 landed
    } else {
      asm volatile("s_wait_asynccnt 0x0" ::: "memory");
    }
    __syncthreads();  // all waves' copies for stage k0 visible

    const bf16_t* laf = &ldsA[cur][0];
    const bf16_t* lbf = &ldsB[cur][0];
    FragBF af[4], bfr[2];
    #pragma unroll
    for (int i = 0; i < 4; ++i) {
      const bf16_t* p = laf + (wm * 64 + i * 16 + lrow) * 32 + lksel;
      af[i].q[0] = *(const v4u*)(p);
      af[i].q[1] = *(const v4u*)(p + 16);
    }
    #pragma unroll
    for (int j = 0; j < 2; ++j) {
      const bf16_t* p = lbf + (wn * 32 + j * 16 + lrow) * 32 + lksel;
      bfr[j].q[0] = *(const v4u*)(p);
      bfr[j].q[1] = *(const v4u*)(p + 16);
    }
    #pragma unroll
    for (int i = 0; i < 4; ++i)
      #pragma unroll
      for (int j = 0; j < 2; ++j)
        acc[i][j] = __builtin_amdgcn_wmma_f32_16x16x32_bf16(
            false, af[i].v, false, bfr[j].v, (short)0, acc[i][j], false, false);

    __syncthreads();  // everyone done reading buf `cur` before it is re-filled
  }

  // C/D layout: VGPR r -> M = (lane>>4)*8 + r, N = lane&15 (within 16x16 frag)
  const int m0 = mBlk + wm * 64;
  const int n0 = nBlk + wn * 32;
  #pragma unroll
  for (int i = 0; i < 4; ++i) {
    #pragma unroll
    for (int j = 0; j < 2; ++j) {
      const int ncol  = n0 + j * 16 + lrow;
      const int mbase = m0 + i * 16 + (lane >> 4) * 8;
      #pragma unroll
      for (int r = 0; r < 8; ++r) {
        float v = acc[i][j][r];
        if (epilogue == 1) {
          v += bias[ncol];
          v = v / (1.f + __expf(-v));   // swish
        }
        D[(long)(mbase + r) * N + ncol] = v;
      }
    }
  }
}

// --------------------------------------------- row L2-normalize -> bf16 ----
__global__ __launch_bounds__(256) void rownorm_bf16(
    const float* __restrict__ X, bf16_t* __restrict__ Y, int K)
{
  const long row = blockIdx.x;
  const float* xr = X + row * (long)K;
  bf16_t* yr = Y + row * (long)K;

  float s = 0.f;
  for (int k = threadIdx.x; k < K; k += 256) { float v = xr[k]; s += v * v; }

  __shared__ float red[256];
  red[threadIdx.x] = s;
  __syncthreads();
  for (int o = 128; o > 0; o >>= 1) {
    if (threadIdx.x < o) red[threadIdx.x] += red[threadIdx.x + o];
    __syncthreads();
  }
  const float rn = rsqrtf(red[0]);
  for (int k = threadIdx.x; k < K; k += 256)
    yr[k] = (bf16_t)(xr[k] * rn);
}

// ----------------------------------------- elementwise f32 -> bf16 ---------
__global__ __launch_bounds__(256) void cvt_f32_bf16(
    const float* __restrict__ in, bf16_t* __restrict__ out, long n)
{
  long i = (long)blockIdx.x * 256 + threadIdx.x;
  if (i < n) out[i] = (bf16_t)in[i];
}

// --------------------------- column-wise online softmax statistics ---------
__global__ __launch_bounds__(256) void colstats(
    const float* __restrict__ C, float* __restrict__ cmax, float* __restrict__ cinv,
    int M, int N, float invT)
{
  const int j = blockIdx.x * 256 + threadIdx.x;
  const int b = blockIdx.y;
  const float* Cb = C + (long)b * M * N;
  float m = -3.0e38f, s = 0.f;
  for (int i = 0; i < M; ++i) {
    float v  = Cb[(long)i * N + j] * invT;
    float nm = fmaxf(m, v);
    s = s * __expf(m - nm) + __expf(v - nm);
    m = nm;
  }
  cmax[(long)b * N + j] = m;
  cinv[(long)b * N + j] = 1.f / s;
}

// ------------------- tiled LDS transpose, f32 -> bf16, optional softmax ----
__global__ __launch_bounds__(1024) void transpose_f32_bf16(
    const float* __restrict__ in, bf16_t* __restrict__ out,
    long inBS, long outBS, int rows, int cols,
    const float* __restrict__ cmax, const float* __restrict__ cinv,
    long statBS, float invT, int mode)
{
  __shared__ float t[32][33];
  const int b = blockIdx.z;
  in  += (long)b * inBS;
  out += (long)b * outBS;
  const int j0 = blockIdx.x * 32, i0 = blockIdx.y * 32;
  const int tx = threadIdx.x, ty = threadIdx.y;

  t[ty][tx] = in[(long)(i0 + ty) * cols + (j0 + tx)];
  __syncthreads();

  const int j = j0 + ty;
  float v = t[tx][ty];                  // = in[i0+tx][j0+ty]
  if (mode == 1)
    v = __expf(v * invT - cmax[(long)b * statBS + j]) * cinv[(long)b * statBS + j];
  out[(long)j * rows + (i0 + tx)] = (bf16_t)v;
}

// ------------------------------------- direct 3x3 SAME conv + swish (+BN) --
__global__ void conv3x3_swish(
    const float* __restrict__ x, const float* __restrict__ w,
    const float* __restrict__ bias,
    const float* __restrict__ bng, const float* __restrict__ bnb,
    float* __restrict__ y, int H, int W, int Cin, int Cout, int sub)
{
  const int wq = blockIdx.x, h = blockIdx.y, b = blockIdx.z;
  const int Hin = H * sub, Win = W * sub;

  for (int co = threadIdx.x; co < Cout; co += blockDim.x) {
    float acc = bias[co];
    #pragma unroll
    for (int dy = -1; dy <= 1; ++dy) {
      const int ih = h + dy;
      if (ih < 0 || ih >= H) continue;
      #pragma unroll
      for (int dx = -1; dx <= 1; ++dx) {
        const int iw = wq + dx;
        if (iw < 0 || iw >= W) continue;
        const float* xp = x + (((long)b * Hin + (long)ih * sub) * Win + (long)iw * sub) * Cin;
        const float* wp = w + ((long)((dy + 1) * 3 + (dx + 1)) * Cin) * Cout + co;
        for (int ci = 0; ci < Cin; ++ci)
          acc = fmaf(xp[ci], wp[(long)ci * Cout], acc);
      }
    }
    float s = acc / (1.f + __expf(-acc));       // swish
    if (bng) s = bng[co] * s * RSQRT_1P001 + bnb[co];
    y[(((long)b * H + h) * W + wq) * Cout + co] = s;
  }
}

// -------------------------------- conf head: sigmoid(hidden . wc2 + bc2) ---
__global__ __launch_bounds__(256) void conf_head(
    const float* __restrict__ hidden, const float* __restrict__ wc2,
    const float* __restrict__ bc2, float* __restrict__ out)
{
  const long row = blockIdx.x;
  const float* hr = hidden + row * 1024;
  float s = 0.f;
  for (int k = threadIdx.x; k < 1024; k += 256) s += hr[k] * wc2[k];
  __shared__ float red[256];
  red[threadIdx.x] = s;
  __syncthreads();
  for (int o = 128; o > 0; o >>= 1) {
    if (threadIdx.x < o) red[threadIdx.x] += red[threadIdx.x + o];
    __syncthreads();
  }
  if (threadIdx.x == 0) {
    float v = red[0] + bc2[0];
    out[row] = 1.f / (1.f + __expf(-v));
  }
}

// ---------------------------------------------------------------- host -----
extern "C" void kernel_launch(void* const* d_in, const int* in_sizes, int n_in,
                              void* d_out, int out_size, void* d_ws, size_t ws_size,
                              hipStream_t stream) {
  (void)in_sizes; (void)n_in; (void)out_size; (void)ws_size;

  const int B = 4, NQ = 4096, KC = 1984;

  const float* feat_tl = (const float*)d_in[0];
  const float* feat_tr = (const float*)d_in[1];
  const float* r_ab    = (const float*)d_in[2];
  const float* w11 = (const float*)d_in[3],  *b11 = (const float*)d_in[4];
  const float* w12 = (const float*)d_in[5],  *b12 = (const float*)d_in[6];
  const float* w21 = (const float*)d_in[7],  *b21 = (const float*)d_in[8];
  const float* w22 = (const float*)d_in[9],  *b22 = (const float*)d_in[10];
  const float* bn1_g = (const float*)d_in[11], *bn1_b = (const float*)d_in[12];
  const float* w31 = (const float*)d_in[13], *b31 = (const float*)d_in[14];
  const float* w32 = (const float*)d_in[15], *b32 = (const float*)d_in[16];
  const float* bn2_g = (const float*)d_in[17], *bn2_b = (const float*)d_in[18];
  const float* w41 = (const float*)d_in[19], *b41 = (const float*)d_in[20];
  const float* w42 = (const float*)d_in[21], *b42 = (const float*)d_in[22];
  const float* bn3_g = (const float*)d_in[23], *bn3_b = (const float*)d_in[24];
  const float* wc1 = (const float*)d_in[25], *bc1 = (const float*)d_in[26];
  const float* wc2 = (const float*)d_in[27], *bc2 = (const float*)d_in[28];

  // ---- workspace carve-up (256B aligned) ----
  char* ws = (char*)d_ws;
  size_t off = 0;
  auto alloc = [&](size_t bytes) {
    void* p = ws + off;
    off += (bytes + 255) & ~(size_t)255;
    return p;
  };
  float*  f1     = (float*) alloc((size_t)B*128*128*64*4);
  float*  f2     = (float*) alloc((size_t)B*128*128*64*4);
  float*  f3     = (float*) alloc((size_t)B*64*64*128*4);
  float*  rabf   = (float*) alloc((size_t)B*64*64*128*4);     // f after BN == rab_flat
  bf16_t* tl_n   = (bf16_t*)alloc((size_t)B*NQ*KC*2);
  bf16_t* tr_n   = (bf16_t*)alloc((size_t)B*NQ*KC*2);
  float*  Cbuf   = (float*) alloc((size_t)B*NQ*NQ*4);
  bf16_t* Cbf    = (bf16_t*)alloc((size_t)B*NQ*NQ*2);         // later reused as P^T
  bf16_t* wc1t   = (bf16_t*)alloc((size_t)1024*NQ*2);
  float*  hidden = (float*) alloc((size_t)B*NQ*1024*4);
  bf16_t* rabt   = (bf16_t*)alloc((size_t)B*128*NQ*2);
  float*  cmaxb  = (float*) alloc((size_t)B*NQ*4);
  float*  cinvb  = (float*) alloc((size_t)B*NQ*4);
  float*  a31    = (float*) alloc((size_t)B*32*32*256*4);
  float*  a41    = (float*) alloc((size_t)B*16*16*512*4);

  // ---- output carve-up ----
  float* out_conf   = (float*)d_out;                      // B*4096
  float* out_align1 = out_conf   + (size_t)B*NQ;          // B*64*64*128
  float* out_align2 = out_align1 + (size_t)B*NQ*128;      // B*32*32*256
  float* out_align3 = out_align2 + (size_t)B*32*32*256;   // B*16*16*512

  // 1) L2-normalize feature rows -> bf16
  rownorm_bf16<<<dim3(B*NQ), 256, 0, stream>>>(feat_tl, tl_n, KC);
  rownorm_bf16<<<dim3(B*NQ), 256, 0, stream>>>(feat_tr, tr_n, KC);

  // 2) conv stack 1: r_ab -> rab_flat (B,64,64,128)
  conv3x3_swish<<<dim3(128,128,B),  64, 0, stream>>>(r_ab, w11, b11, nullptr, nullptr, f1, 128,128,  2, 64, 1);
  conv3x3_swish<<<dim3(128,128,B),  64, 0, stream>>>(f1,   w12, b12, nullptr, nullptr, f2, 128,128, 64, 64, 1);
  conv3x3_swish<<<dim3( 64, 64,B), 128, 0, stream>>>(f2,   w21, b21, nullptr, nullptr, f3,  64, 64, 64,128, 2);
  conv3x3_swish<<<dim3( 64, 64,B), 128, 0, stream>>>(f3,   w22, b22, bn1_g,  bn1_b,   rabf, 64, 64,128,128, 1);

  // 3) correlation GEMM: C[b,i,j] = sum_c tr_n[b,i,c]*tl_n[b,j,c]
  gemm_bf16_wmma<<<dim3(NQ/128, NQ/128, B), 256, 0, stream>>>(
      tr_n, (long)NQ*KC, tl_n, (long)NQ*KC, Cbuf, (long)NQ*NQ,
      NQ, NQ, KC, nullptr, 0);

  // 4) C -> bf16; wc1 -> wc1^T bf16
  {
    long n = (long)B*NQ*NQ;
    cvt_f32_bf16<<<dim3((unsigned)((n + 255) / 256)), 256, 0, stream>>>(Cbuf, Cbf, n);
  }
  transpose_f32_bf16<<<dim3(1024/32, NQ/32, 1), dim3(32,32), 0, stream>>>(
      wc1, wc1t, 0, 0, NQ, 1024, nullptr, nullptr, 0, 0.f, 0);

  // 5) hidden = swish(C @ wc1 + bc1)   (M=4096, N=1024, K=4096, batched)
  gemm_bf16_wmma<<<dim3(1024/128, NQ/128, B), 256, 0, stream>>>(
      Cbf, (long)NQ*NQ, wc1t, 0, hidden, (long)NQ*1024,
      NQ, 1024, NQ, bc1, 1);

  // 6) conf = sigmoid(hidden @ wc2 + bc2)
  conf_head<<<dim3(B*NQ), 256, 0, stream>>>(hidden, wc2, bc2, out_conf);

  // 7) softmax over axis=1 (columns of C), write P^T in bf16 (reuse Cbf)
  colstats<<<dim3(NQ/256, B), 256, 0, stream>>>(Cbuf, cmaxb, cinvb, NQ, NQ, 100.0f);
  transpose_f32_bf16<<<dim3(NQ/32, NQ/32, B), dim3(32,32), 0, stream>>>(
      Cbuf, Cbf, (long)NQ*NQ, (long)NQ*NQ, NQ, NQ, cmaxb, cinvb, NQ, 100.0f, 1);

  // 8) rab_flat -> rab^T bf16 (B,128,4096)
  transpose_f32_bf16<<<dim3(128/32, NQ/32, B), dim3(32,32), 0, stream>>>(
      rabf, rabt, (long)NQ*128, (long)128*NQ, NQ, 128, nullptr, nullptr, 0, 0.f, 0);

  // 9) align_1 = P^T @ rab   (M=4096 j, N=128 c, K=4096 i)
  gemm_bf16_wmma<<<dim3(128/128, NQ/128, B), 256, 0, stream>>>(
      Cbf, (long)NQ*NQ, rabt, (long)128*NQ, out_align1, (long)NQ*128,
      NQ, 128, NQ, nullptr, 0);

  // 10) conv stack 2/3: align_2, align_3
  conv3x3_swish<<<dim3(32,32,B), 256, 0, stream>>>(out_align1, w31, b31, nullptr, nullptr, a31, 32,32,128,256, 2);
  conv3x3_swish<<<dim3(32,32,B), 256, 0, stream>>>(a31,        w32, b32, bn2_g,  bn2_b, out_align2, 32,32,256,256, 1);
  conv3x3_swish<<<dim3(16,16,B), 256, 0, stream>>>(out_align2, w41, b41, nullptr, nullptr, a41, 16,16,256,512, 2);
  conv3x3_swish<<<dim3(16,16,B), 256, 0, stream>>>(a41,        w42, b42, bn3_g,  bn3_b, out_align3, 16,16,512,512, 1);
}
